// ROIHeads_73701638800034
// MI455X (gfx1250) — matrix operand order repaired and checked by reading, not compiled, not run on hardware
//
#include <hip/hip_runtime.h>
#include <hip/hip_bf16.h>

// ---------------- types ----------------
typedef __attribute__((ext_vector_type(8)))  __bf16 v8bf;
typedef __attribute__((ext_vector_type(16))) __bf16 v16bf;
typedef __attribute__((ext_vector_type(8)))  float  v8f;

#define IMG_SZ     1024.0f
#define OUT_P      7
#define SCALE_F    (1.0f/32.0f)
#define DWH_CLIP_F 4.135166556742356f   // log(1000/16)

// ---------------- fp32 -> bf16 convert ----------------
__global__ void cvt_f32_bf16(const float* __restrict__ src, __bf16* __restrict__ dst, int n) {
    int i = blockIdx.x * blockDim.x + threadIdx.x;
    if (i < n) dst[i] = (__bf16)src[i];
}

// ---------------- ROIAlign: features[2,256,32,32] + proposals[2,1024,4] -> pooled bf16 [2048][12544]
__global__ void roi_align_kernel(const float* __restrict__ feat,
                                 const float* __restrict__ props,
                                 __bf16* __restrict__ pooled) {
    const long long idx = (long long)blockIdx.x * blockDim.x + threadIdx.x;
    if (idx >= 2048LL * 256 * 49) return;
    const int p   = (int)(idx % 49);
    const int tmp = (int)(idx / 49);
    const int c   = tmp & 255;          // % 256
    const int m   = tmp >> 8;           // / 256  (0..2047)
    const int oy  = p / 7, ox = p % 7;
    const int b   = m >> 10;            // image index

    const float* roi = props + (size_t)m * 4;
    const float x1 = roi[0] * SCALE_F;
    const float y1 = roi[1] * SCALE_F;
    const float rw = fmaxf(roi[2] * SCALE_F - x1, 1.0f);
    const float rh = fmaxf(roi[3] * SCALE_F - y1, 1.0f);
    const float bw = rw * (1.0f / OUT_P);
    const float bh = rh * (1.0f / OUT_P);

    const float* f = feat + ((size_t)(b * 256 + c)) * 1024;  // 32x32 plane

    float sum = 0.0f;
    #pragma unroll
    for (int iy = 0; iy < 2; ++iy) {
        const float gy = (float)oy + ((float)iy + 0.5f) * 0.5f;
        const float y  = y1 + gy * bh;
        #pragma unroll
        for (int ix = 0; ix < 2; ++ix) {
            const float gx = (float)ox + ((float)ix + 0.5f) * 0.5f;
            const float x  = x1 + gx * bw;
            const bool valid = (y > -1.0f) && (y < 32.0f) && (x > -1.0f) && (x < 32.0f);
            if (!valid) continue;
            float yc = fminf(fmaxf(y, 0.0f), 31.0f);
            float xc = fminf(fmaxf(x, 0.0f), 31.0f);
            int y0 = (int)floorf(yc);
            int x0 = (int)floorf(xc);
            int y1i = min(y0 + 1, 31);
            int x1i = min(x0 + 1, 31);
            float ly = yc - (float)y0, lx = xc - (float)x0;
            float hy = 1.0f - ly,      hx = 1.0f - lx;
            sum += f[y0 * 32 + x0] * (hy * hx) + f[y0 * 32 + x1i] * (hy * lx)
                 + f[y1i * 32 + x0] * (ly * hx) + f[y1i * 32 + x1i] * (ly * lx);
        }
    }
    pooled[(size_t)m * 12544 + c * 49 + p] = (__bf16)(sum * 0.25f);
}

// ---------------- bf16 WMMA GEMM: Out[M][N] = relu(A[M][K] * B[N][K]^T + bias) ----------------
// Block = 64 threads = 2 wave32 (side by side in N).  Block tile 64x128;
// wave tile 64(M)x64(N) = 4x4 WMMA tiles -> 16 WMMA per 8 fragment loads (32.8 flop/L2-byte).
// Software-pipelined with register double-buffering (NSTEP must be even).

template<int KDIM>
__device__ __forceinline__ void load_frags(const __bf16* __restrict__ paBase0,
                                           const __bf16* __restrict__ pbBase0,
                                           int k0, v16bf a[4], v16bf b[4]) {
    #pragma unroll
    for (int i = 0; i < 4; ++i) {
        // A layout (matches B): lanes 0-15 hold K 0..15 ; lanes 16-31 hold K 16..31
        const __bf16* pa = paBase0 + (size_t)i * 16 * KDIM + k0;
        v8bf lo = *(const v8bf*)pa;
        v8bf hi = *(const v8bf*)(pa + 8);
        v16bf av;
        #pragma unroll
        for (int t = 0; t < 8; ++t) { av[t] = lo[t]; av[8 + t] = hi[t]; }
        a[i] = av;
        __builtin_prefetch(pa + 128, 0, 3);   // 2 pipeline stages ahead -> global_prefetch_b8
    }
    #pragma unroll
    for (int j = 0; j < 4; ++j) {
        const __bf16* pb = pbBase0 + (size_t)j * 16 * KDIM + k0;
        b[j] = *(const v16bf*)pb;             // 16 contiguous K elements per lane
        __builtin_prefetch(pb + 128, 0, 3);
    }
}

__device__ __forceinline__ void mma_frags(v8f acc[4][4], const v16bf a[4], const v16bf b[4]) {
    #pragma unroll
    for (int i = 0; i < 4; ++i)
        #pragma unroll
        for (int j = 0; j < 4; ++j)
            acc[i][j] = __builtin_amdgcn_wmma_f32_16x16x32_bf16(
                false, a[i], false, b[j], (short)0, acc[i][j], false, false);
}

template<int KDIM>
__global__ __launch_bounds__(64)
void gemm_bf16_relu(const __bf16* __restrict__ A,   // [M][KDIM] row-major
                    const __bf16* __restrict__ B,   // [N][KDIM] row-major (weights)
                    const float*  __restrict__ bias,// [N]
                    __bf16* __restrict__ Out,       // [M][N]
                    int Ncols) {
    const int tid  = threadIdx.x;
    const int lane = tid & 31;
    const int wave = tid >> 5;        // 0..1
    const int r    = lane & 15;
    const int half = lane >> 4;

    const int mBase = blockIdx.x * 64;
    const int nBase = blockIdx.y * 128 + wave * 64;

    // per-lane base pointers (k-invariant part)
    const __bf16* paBase0 = A + (size_t)(mBase + r) * KDIM + half * 16;
    const __bf16* pbBase0 = B + (size_t)(nBase + r) * KDIM + half * 16;

    v8f acc[4][4];
    #pragma unroll
    for (int i = 0; i < 4; ++i)
        #pragma unroll
        for (int j = 0; j < 4; ++j)
            acc[i][j] = (v8f)(0.0f);

    constexpr int NSTEP = KDIM / 32;  // even for both 12544 (392) and 1024 (32)
    v16bf a0[4], b0[4], a1[4], b1[4];

    load_frags<KDIM>(paBase0, pbBase0, 0, a0, b0);
    #pragma unroll 1
    for (int s = 0; s < NSTEP - 2; s += 2) {
        load_frags<KDIM>(paBase0, pbBase0, (s + 1) * 32, a1, b1);
        mma_frags(acc, a0, b0);
        load_frags<KDIM>(paBase0, pbBase0, (s + 2) * 32, a0, b0);
        mma_frags(acc, a1, b1);
    }
    load_frags<KDIM>(paBase0, pbBase0, (NSTEP - 1) * 32, a1, b1);
    mma_frags(acc, a0, b0);
    mma_frags(acc, a1, b1);

    // ---- epilogue: C/D layout -> element (vgpr v, lane L): M = v + 8*(L>>4), N = L&15
    #pragma unroll
    for (int j = 0; j < 4; ++j) {
        const int col = nBase + j * 16 + r;
        const float bv = bias[col];
        #pragma unroll
        for (int i = 0; i < 4; ++i) {
            #pragma unroll
            for (int v = 0; v < 8; ++v) {
                const int row = mBase + i * 16 + v + 8 * half;
                float val = fmaxf(acc[i][j][v] + bv, 0.0f);
                Out[(size_t)row * Ncols + col] = (__bf16)val;
            }
        }
    }
}

// ---------------- small heads: cls (2) + bbox (8) per row ----------------
__global__ void heads_kernel(const __bf16* __restrict__ x,   // [2048][1024]
                             const float* __restrict__ wc, const float* __restrict__ bc,
                             const float* __restrict__ wb, const float* __restrict__ bb,
                             float* __restrict__ out_cls,    // [2048][2]
                             float* __restrict__ out_del) {  // [2048][8]
    const int t = blockIdx.x * blockDim.x + threadIdx.x;     // 2048*10 threads
    if (t >= 2048 * 10) return;
    const int j = t % 10;
    const int m = t / 10;
    const float* w = (j < 2) ? (wc + (size_t)j * 1024) : (wb + (size_t)(j - 2) * 1024);
    float acc = (j < 2) ? bc[j] : bb[j - 2];
    const __bf16* xr = x + (size_t)m * 1024;
    #pragma unroll 1
    for (int k = 0; k < 1024; k += 8) {
        v8bf xv = *(const v8bf*)(xr + k);
        #pragma unroll
        for (int t8 = 0; t8 < 8; ++t8)
            acc += (float)xv[t8] * w[k + t8];
    }
    if (j < 2) out_cls[m * 2 + j] = acc;
    else       out_del[m * 8 + (j - 2)] = acc;
}

// ---------------- box decode + softmax ----------------
__global__ void decode_kernel(const float* __restrict__ props,   // [2048][4]
                              const float* __restrict__ cls,     // [2048][2]
                              const float* __restrict__ deltas,  // [2048][8]
                              float* __restrict__ boxes,         // [2048][2][4]
                              float* __restrict__ scores) {      // [2048][2]
    const int m = blockIdx.x * blockDim.x + threadIdx.x;
    if (m >= 2048) return;
    const float* pr = props + (size_t)m * 4;
    const float pw = pr[2] - pr[0];
    const float ph = pr[3] - pr[1];
    const float cx = pr[0] + 0.5f * pw;
    const float cy = pr[1] + 0.5f * ph;
    #pragma unroll
    for (int k = 0; k < 2; ++k) {
        const float* d = deltas + (size_t)m * 8 + k * 4;
        const float dx = d[0] * 0.1f;                       // / wx(10)
        const float dy = d[1] * 0.1f;                       // / wy(10)
        const float dw = fminf(d[2] * 0.2f, DWH_CLIP_F);    // / ww(5)
        const float dh = fminf(d[3] * 0.2f, DWH_CLIP_F);    // / wh(5)
        const float pcx = dx * pw + cx;
        const float pcy = dy * ph + cy;
        const float ppw = expf(dw) * pw;
        const float pph = expf(dh) * ph;
        float bx0 = pcx - 0.5f * ppw, by0 = pcy - 0.5f * pph;
        float bx1 = pcx + 0.5f * ppw, by1 = pcy + 0.5f * pph;
        float* o = boxes + (size_t)m * 8 + k * 4;
        o[0] = fminf(fmaxf(bx0, 0.0f), IMG_SZ);
        o[1] = fminf(fmaxf(by0, 0.0f), IMG_SZ);
        o[2] = fminf(fmaxf(bx1, 0.0f), IMG_SZ);
        o[3] = fminf(fmaxf(by1, 0.0f), IMG_SZ);
    }
    const float s0 = cls[m * 2 + 0], s1 = cls[m * 2 + 1];
    const float mx = fmaxf(s0, s1);
    const float e0 = expf(s0 - mx), e1 = expf(s1 - mx);
    const float inv = 1.0f / (e0 + e1);
    scores[m * 2 + 0] = e0 * inv;
    scores[m * 2 + 1] = e1 * inv;
}

// ---------------- launch ----------------
extern "C" void kernel_launch(void* const* d_in, const int* in_sizes, int n_in,
                              void* d_out, int out_size, void* d_ws, size_t ws_size,
                              hipStream_t stream) {
    const float* features  = (const float*)d_in[0]; // [2,256,32,32]
    const float* proposals = (const float*)d_in[1]; // [2,1024,4]
    const float* w1 = (const float*)d_in[2];        // [1024,12544]
    const float* b1 = (const float*)d_in[3];        // [1024]
    const float* w2 = (const float*)d_in[4];        // [1024,1024]
    const float* b2 = (const float*)d_in[5];        // [1024]
    const float* wc = (const float*)d_in[6];        // [2,1024]
    const float* bc = (const float*)d_in[7];        // [2]
    const float* wb = (const float*)d_in[8];        // [8,1024]
    const float* bb = (const float*)d_in[9];        // [8]

    char* ws = (char*)d_ws;
    size_t off = 0;
    __bf16* w1b    = (__bf16*)(ws + off); off += (size_t)12544 * 1024 * 2;   // 25.7 MB
    __bf16* w2b    = (__bf16*)(ws + off); off += (size_t)1024  * 1024 * 2;   //  2.1 MB
    __bf16* pooled = (__bf16*)(ws + off); off += (size_t)2048  * 12544 * 2;  // 51.4 MB
    __bf16* x1     = (__bf16*)(ws + off); off += (size_t)2048  * 1024 * 2;   //  4.2 MB
    __bf16* x2     = (__bf16*)(ws + off); off += (size_t)2048  * 1024 * 2;   //  4.2 MB

    float* out       = (float*)d_out;
    float* out_cls   = out;               // [2048,2]
    float* out_del   = out + 4096;        // [2048,8]
    float* out_boxes = out + 20480;       // [2048,2,4]
    float* out_score = out + 36864;       // [2048,2]

    // 1) weight conversion fp32 -> bf16
    {
        int n1 = 12544 * 1024;
        cvt_f32_bf16<<<(n1 + 255) / 256, 256, 0, stream>>>(w1, w1b, n1);
        int n2 = 1024 * 1024;
        cvt_f32_bf16<<<(n2 + 255) / 256, 256, 0, stream>>>(w2, w2b, n2);
    }
    // 2) ROI align -> pooled bf16 [2048][12544]
    {
        long long total = 2048LL * 256 * 49;
        int blocks = (int)((total + 255) / 256);   // 100352
        roi_align_kernel<<<blocks, 256, 0, stream>>>(features, proposals, pooled);
    }
    // 3) FC1: [2048 x 12544] * [12544 x 1024] -> relu -> x1   (grid 32x8 = 256 WGs x 2 waves)
    gemm_bf16_relu<12544><<<dim3(32, 8), 64, 0, stream>>>(pooled, w1b, b1, x1, 1024);
    // 4) FC2: [2048 x 1024] * [1024 x 1024] -> relu -> x2
    gemm_bf16_relu<1024><<<dim3(32, 8), 64, 0, stream>>>(x1, w2b, b2, x2, 1024);
    // 5) heads
    heads_kernel<<<80, 256, 0, stream>>>(x2, wc, bc, wb, bb, out_cls, out_del);
    // 6) decode + softmax
    decode_kernel<<<8, 256, 0, stream>>>(proposals, out_cls, out_del, out_boxes, out_score);
}
